// HGT_68753836474875
// MI455X (gfx1250) — matrix operand order repaired and checked by reading, not compiled, not run on hardware
//
#include <hip/hip_runtime.h>

typedef __attribute__((ext_vector_type(16))) __bf16 v16bf;
typedef __attribute__((ext_vector_type(8)))  __bf16 v8bf;
typedef __attribute__((ext_vector_type(8)))  float  v8f;

constexpr int kNC = 60000, kNV = 256, kC = 128, kH = 4, kD = 32, kL = 3;
constexpr int kE1 = 600000, kE2 = 240000;

__device__ __forceinline__ float gelu_t(float x) {
  float x3 = x * x * x;
  return 0.5f * x * (1.0f + tanhf(0.7978845608028654f * (x + 0.044715f * x3)));
}
__device__ __forceinline__ unsigned fenc(float f) {
  unsigned u = __float_as_uint(f);
  return (u & 0x80000000u) ? ~u : (u | 0x80000000u);
}
__device__ __forceinline__ float fdec(unsigned u) {
  unsigned b = (u & 0x80000000u) ? (u ^ 0x80000000u) : ~u;
  return __uint_as_float(b);
}

// ---------------- utility kernels ----------------
__global__ void fill_f32(float* p, float v, int n) {
  int i = blockIdx.x * 256 + threadIdx.x;
  if (i < n) p[i] = v;
}
__global__ void fill_u32(unsigned* p, unsigned v, int n) {
  int i = blockIdx.x * 256 + threadIdx.x;
  if (i < n) p[i] = v;
}
__global__ void cvt_bf16(const float* __restrict__ x, __bf16* __restrict__ y, int n) {
  int i = blockIdx.x * 256 + threadIdx.x;
  if (i < n) y[i] = (__bf16)x[i];
}
__global__ void gelu_cvt_bf16(const float* __restrict__ x, __bf16* __restrict__ y, int n) {
  int i = blockIdx.x * 256 + threadIdx.x;
  if (i < n) y[i] = (__bf16)gelu_t(x[i]);
}
// Wt[n][k] = bf16(W[k][n])  (pre-transpose B for contiguous WMMA fragment loads)
__global__ void transpose_cvt(const float* __restrict__ W, __bf16* __restrict__ Wt) {
  int kk = blockIdx.x, n = threadIdx.x;
  Wt[n * kC + kk] = (__bf16)W[kk * kC + n];
}
// Fold per-head relation matrix into projection weight:
// Wt[h*32+e][k] = sum_d W[k][h*32+d] * rel[h][d][e];  bOut likewise from bias.
__global__ void fold_weight(const float* __restrict__ W, const float* __restrict__ rel,
                            const float* __restrict__ b, __bf16* __restrict__ Wt,
                            float* __restrict__ bOut) {
  int kk = blockIdx.x, n = threadIdx.x;
  int h = n >> 5, e = n & 31;
  const float* Wr = W + kk * kC + h * kD;
  const float* rr = rel + h * kD * kD;
  float acc = 0.f;
#pragma unroll
  for (int d = 0; d < kD; ++d) acc += Wr[d] * rr[d * kD + e];
  Wt[n * kC + kk] = (__bf16)acc;
  if (kk == 0) {
    float bb = 0.f;
    for (int d = 0; d < kD; ++d) bb += b[h * kD + d] * rr[d * kD + e];
    bOut[n] = bb;
  }
}

// ---------------- WMMA GEMM: out[N x 128] = A(bf16)[N x 128] @ Bt^T + bias ----------------
// Bt is stored N-major (Bt[n][k]).  mode 0: out = acc + bias
// mode 1: out = sig(skip)*(acc+bias) + (1-sig(skip))*xold  (skip connection fused)
__global__ void gemm128_wmma(const __bf16* __restrict__ A, const __bf16* __restrict__ Bt,
                             const float* __restrict__ bias, float* __restrict__ out,
                             const float* __restrict__ xold, const float* __restrict__ skipPtr,
                             int mode) {
  int lane = threadIdx.x & 31;
  int wv = threadIdx.x >> 5;           // 8 waves, each owns 16 cols
  int row0 = blockIdx.x * 16;
  int col0 = wv * 16;
  int hf = lane >> 4;                  // lane half selects K sub-block per ISA layout
  int l15 = lane & 15;
  const __bf16* arow = A + (size_t)(row0 + l15) * kC + hf * 8;
  const __bf16* brow = Bt + (size_t)(col0 + l15) * kC + hf * 8;
  v8f acc = {};
#pragma unroll
  for (int kt = 0; kt < kC; kt += 32) {
    v8bf a0 = *(const v8bf*)(arow + kt);        // K: base..base+7
    v8bf a1 = *(const v8bf*)(arow + kt + 16);   // K: base+16..base+23
    v8bf b0 = *(const v8bf*)(brow + kt);        // K: base..base+7 (col-major)
    v8bf b1 = *(const v8bf*)(brow + kt + 16);   // K: base+16..base+23
    v16bf av = __builtin_shufflevector(a0, a1, 0, 1, 2, 3, 4, 5, 6, 7, 8, 9, 10, 11, 12, 13, 14, 15);
    v16bf bvv = __builtin_shufflevector(b0, b1, 0, 1, 2, 3, 4, 5, 6, 7, 8, 9, 10, 11, 12, 13, 14, 15);
    acc = __builtin_amdgcn_wmma_f32_16x16x32_bf16(false, av, false, bvv, (short)0, acc, false, false);
  }
  int col = col0 + l15;
  float bs = bias ? bias[col] : 0.f;
  if (mode == 0) {
#pragma unroll
    for (int j = 0; j < 8; ++j)
      out[(size_t)(row0 + j + hf * 8) * kC + col] = acc[j] + bs;
  } else {
    float sg = 1.0f / (1.0f + __expf(-skipPtr[0]));
#pragma unroll
    for (int j = 0; j < 8; ++j) {
      size_t idx = (size_t)(row0 + j + hf * 8) * kC + col;
      out[idx] = sg * (acc[j] + bs) + (1.0f - sg) * xold[idx];
    }
  }
}

// ---------------- edge attention passes (H=4, D=32), one wave32 per edge ----------------
__global__ void edge_logits_max(const float* __restrict__ q, const float* __restrict__ k,
                                const int* __restrict__ src, const int* __restrict__ dst,
                                const float* __restrict__ prel, float* __restrict__ elog,
                                unsigned* __restrict__ menc, int E) {
  int gid = blockIdx.x * 256 + threadIdx.x;
  int e = gid >> 5;
  if (e >= E) return;
  int lane = threadIdx.x & 31;
  int s = src[e], d = dst[e];
  float4 q4 = ((const float4*)q)[(size_t)d * 32 + lane];
  float4 k4 = ((const float4*)k)[(size_t)s * 32 + lane];
  float p = q4.x * k4.x + q4.y * k4.y + q4.z * k4.z + q4.w * k4.w;
  p += __shfl_xor(p, 1, 32);
  p += __shfl_xor(p, 2, 32);
  p += __shfl_xor(p, 4, 32);           // sum over each 8-lane head group
  int h = lane >> 3;
  if ((lane & 7) == 0) {
    float lg = p * prel[h] * 0.17677669529663687f;  // 1/sqrt(32)
    elog[(size_t)e * 4 + h] = lg;
    atomicMax(&menc[(size_t)d * 4 + h], fenc(lg));
  }
}
__global__ void edge_exp_sum(const int* __restrict__ dst, const unsigned* __restrict__ menc,
                             float* __restrict__ elog, float* __restrict__ ssum, int EH) {
  int i = blockIdx.x * 256 + threadIdx.x;
  if (i >= EH) return;
  int e = i >> 2, h = i & 3;
  int d = dst[e];
  float m = fdec(menc[(size_t)d * 4 + h]);
  float ex = __expf(elog[i] - m);
  elog[i] = ex;
  atomicAdd(&ssum[(size_t)d * 4 + h], ex);
}
__global__ void edge_scatter(const float* __restrict__ vrel, const float* __restrict__ elog,
                             const float* __restrict__ ssum, const int* __restrict__ src,
                             const int* __restrict__ dst, float* __restrict__ agg, int E) {
  int gid = blockIdx.x * 256 + threadIdx.x;
  int e = gid >> 5;
  if (e >= E) return;
  int lane = threadIdx.x & 31;
  int s = src[e], d = dst[e];
  int h = lane >> 3;
  float al = elog[(size_t)e * 4 + h] / ssum[(size_t)d * 4 + h];
  float4 v4 = ((const float4*)vrel)[(size_t)s * 32 + lane];
  float* o = agg + (size_t)d * kC + lane * 4;
  atomicAdd(o + 0, al * v4.x);
  atomicAdd(o + 1, al * v4.y);
  atomicAdd(o + 2, al * v4.z);
  atomicAdd(o + 3, al * v4.w);
}

// ---------------- out-conv (H=1, D=1) ----------------
// one wave per node: two simultaneous 128-dot-products
__global__ void node_dot2(const float* __restrict__ x, const float* __restrict__ w1,
                          const float* __restrict__ b1p, const float* __restrict__ w2,
                          const float* __restrict__ b2p, float* __restrict__ o1,
                          float* __restrict__ o2, int n) {
  int node = (blockIdx.x * 256 + threadIdx.x) >> 5;
  if (node >= n) return;
  int lane = threadIdx.x & 31;
  float4 xv4 = ((const float4*)x)[(size_t)node * 32 + lane];
  float4 a4 = ((const float4*)w1)[lane];
  float p1 = xv4.x * a4.x + xv4.y * a4.y + xv4.z * a4.z + xv4.w * a4.w;
  float p2 = 0.f;
  if (w2) {
    float4 c4 = ((const float4*)w2)[lane];
    p2 = xv4.x * c4.x + xv4.y * c4.y + xv4.z * c4.z + xv4.w * c4.w;
  }
#pragma unroll
  for (int m = 1; m < 32; m <<= 1) {
    p1 += __shfl_xor(p1, m, 32);
    p2 += __shfl_xor(p2, m, 32);
  }
  if (lane == 0) {
    o1[node] = p1 + b1p[0];
    if (o2) o2[node] = p2 + b2p[0];
  }
}
__global__ void oedge_logits_max(const float* __restrict__ qo, const float* __restrict__ ko,
                                 const int* __restrict__ src, const int* __restrict__ dst,
                                 const float* __restrict__ arel_o, const float* __restrict__ prel_o,
                                 int li, float* __restrict__ elog, unsigned* __restrict__ menc, int E) {
  int e = blockIdx.x * 256 + threadIdx.x;
  if (e >= E) return;
  float lg = qo[dst[e]] * ko[src[e]] * arel_o[li] * prel_o[li];
  elog[e] = lg;
  atomicMax(&menc[dst[e]], fenc(lg));
}
__global__ void oedge_exp_sum(const int* __restrict__ dst, const unsigned* __restrict__ menc,
                              float* __restrict__ elog, float* __restrict__ ssum, int E) {
  int e = blockIdx.x * 256 + threadIdx.x;
  if (e >= E) return;
  int d = dst[e];
  float ex = __expf(elog[e] - fdec(menc[d]));
  elog[e] = ex;
  atomicAdd(&ssum[d], ex);
}
__global__ void oedge_scatter(const float* __restrict__ vo, const float* __restrict__ elog,
                              const float* __restrict__ ssum, const float* __restrict__ mrel_o,
                              int li, const int* __restrict__ src, const int* __restrict__ dst,
                              float* __restrict__ aggo, int E) {
  int e = blockIdx.x * 256 + threadIdx.x;
  if (e >= E) return;
  int d = dst[e];
  atomicAdd(&aggo[d], (elog[e] / ssum[d]) * vo[src[e]] * mrel_o[li]);
}
__global__ void outs_k(const float* __restrict__ aggo, const float* __restrict__ Wa_o,
                       const float* __restrict__ ba_o, int li, float* __restrict__ outs) {
  int v = threadIdx.x;
  if (v < kNV) outs[li * kNV + v] = gelu_t(aggo[v]) * Wa_o[li] + ba_o[li];
}
__global__ void final_k(const float* __restrict__ outs, const float* __restrict__ fcW,
                        const float* __restrict__ fcb, const float* __restrict__ W1,
                        const float* __restrict__ b1, const float* __restrict__ W2,
                        const float* __restrict__ b2, float* __restrict__ out) {
  int b = threadIdx.x;
  if (b >= 64) return;
  float hsum = fcb[0];
#pragma unroll
  for (int j = 0; j < 4; ++j) {
    float m = outs[b * 4 + j];
    m = fmaxf(m, outs[kNV + b * 4 + j]);
    m = fmaxf(m, outs[2 * kNV + b * 4 + j]);
    hsum += m * fcW[j];
  }
  float g = gelu_t(hsum);
  float h1 = gelu_t(g * W1[0] + b1[0]);
  float h2 = gelu_t(g * W1[1] + b1[1]);
  out[b] = h1 * W2[0] + h2 * W2[1] + b2[0];
}

// ---------------- launch ----------------
extern "C" void kernel_launch(void* const* d_in, const int* in_sizes, int n_in,
                              void* d_out, int out_size, void* d_ws, size_t ws_size,
                              hipStream_t stream) {
  (void)in_sizes; (void)n_in; (void)out_size; (void)ws_size;
  const float* x_cell = (const float*)d_in[0];
  const float* x_vcell = (const float*)d_in[1];
  const float* Wk = (const float*)d_in[2];
  const float* bk = (const float*)d_in[3];
  const float* Wq = (const float*)d_in[4];
  const float* bq = (const float*)d_in[5];
  const float* Wv = (const float*)d_in[6];
  const float* bv = (const float*)d_in[7];
  const float* Wa = (const float*)d_in[8];
  const float* ba = (const float*)d_in[9];
  const float* skip = (const float*)d_in[10];
  const float* arel = (const float*)d_in[11];
  const float* mrel = (const float*)d_in[12];
  const float* prel = (const float*)d_in[13];
  const float* Wk_o = (const float*)d_in[14];
  const float* bk_o = (const float*)d_in[15];
  const float* Wq_o = (const float*)d_in[16];
  const float* bq_o = (const float*)d_in[17];
  const float* Wv_o = (const float*)d_in[18];
  const float* bv_o = (const float*)d_in[19];
  const float* Wa_o = (const float*)d_in[20];
  const float* ba_o = (const float*)d_in[21];
  const float* arel_o = (const float*)d_in[22];
  const float* mrel_o = (const float*)d_in[23];
  const float* prel_o = (const float*)d_in[24];
  const float* fc_W = (const float*)d_in[25];
  const float* fc_b = (const float*)d_in[26];
  const float* mlp_W1 = (const float*)d_in[27];
  const float* mlp_b1 = (const float*)d_in[28];
  const float* mlp_W2 = (const float*)d_in[29];
  const float* mlp_b2 = (const float*)d_in[30];
  const int* src_cc = (const int*)d_in[31];
  const int* dst_cc = (const int*)d_in[32];
  const int* src_cv = (const int*)d_in[33];
  const int* dst_cv = (const int*)d_in[34];

  char* base = (char*)d_ws;
  size_t off = 0;
  auto alloc = [&](size_t bytes) -> void* {
    void* p = base + off;
    off = (off + bytes + 255) & ~(size_t)255;
    return p;
  };
  float* xc = (float*)alloc((size_t)kNC * kC * 4);
  float* xvv = (float*)alloc((size_t)kNV * kC * 4);
  __bf16* xcb = (__bf16*)alloc((size_t)kNC * kC * 2);
  __bf16* xvb = (__bf16*)alloc((size_t)kNV * kC * 2);
  float* q = (float*)alloc((size_t)kNC * kC * 4);
  float* qv = (float*)alloc((size_t)kNV * kC * 4);
  float* krel = (float*)alloc((size_t)kNC * kC * 4);
  float* vrel = (float*)alloc((size_t)kNC * kC * 4);
  float* agg = (float*)alloc((size_t)kNC * kC * 4);
  float* aggv = (float*)alloc((size_t)kNV * kC * 4);
  unsigned* menc = (unsigned*)alloc((size_t)kNC * kH * 4);
  float* ssum = (float*)alloc((size_t)kNC * kH * 4);
  unsigned* mvenc = (unsigned*)alloc((size_t)kNV * kH * 4);
  float* svsum = (float*)alloc((size_t)kNV * kH * 4);
  float* elog = (float*)alloc((size_t)kE1 * kH * 4);
  __bf16* WtQ0 = (__bf16*)alloc((size_t)kC * kC * 2);
  __bf16* WtQ1 = (__bf16*)alloc((size_t)kC * kC * 2);
  __bf16* WtK = (__bf16*)alloc((size_t)kC * kC * 2);
  __bf16* WtV = (__bf16*)alloc((size_t)kC * kC * 2);
  __bf16* WtA0 = (__bf16*)alloc((size_t)kC * kC * 2);
  __bf16* WtA1 = (__bf16*)alloc((size_t)kC * kC * 2);
  float* bKf = (float*)alloc(kC * 4);
  float* bVf = (float*)alloc(kC * 4);
  float* ko = (float*)alloc((size_t)kNC * 4);
  float* vo = (float*)alloc((size_t)kNC * 4);
  float* qo = (float*)alloc((size_t)kNV * 4);
  unsigned* moenc = (unsigned*)alloc((size_t)kNV * 4);
  float* sosum = (float*)alloc((size_t)kNV * 4);
  float* aggo = (float*)alloc((size_t)kNV * 4);
  float* elogo = (float*)alloc((size_t)kE2 * 4);
  float* outsBuf = (float*)alloc((size_t)kL * kNV * 4);

  hipMemcpyAsync(xc, x_cell, (size_t)kNC * kC * 4, hipMemcpyDeviceToDevice, stream);
  hipMemcpyAsync(xvv, x_vcell, (size_t)kNV * kC * 4, hipMemcpyDeviceToDevice, stream);

  auto nb = [](long n) { return (unsigned)((n + 255) / 256); };
  const size_t CC = (size_t)kC * kC;
  const size_t HDD = (size_t)kH * kD * kD;

  for (int li = 0; li < kL; ++li) {
    cvt_bf16<<<nb((long)kNC * kC), 256, 0, stream>>>(xc, xcb, kNC * kC);
    cvt_bf16<<<nb((long)kNV * kC), 256, 0, stream>>>(xvv, xvb, kNV * kC);
    transpose_cvt<<<128, 128, 0, stream>>>(Wq + (li * 2 + 0) * CC, WtQ0);
    transpose_cvt<<<128, 128, 0, stream>>>(Wq + (li * 2 + 1) * CC, WtQ1);
    fold_weight<<<128, 128, 0, stream>>>(Wk + (li * 2 + 0) * CC, arel + (li * 2 + 0) * HDD,
                                         bk + (li * 2 + 0) * kC, WtK, bKf);
    fold_weight<<<128, 128, 0, stream>>>(Wv + (li * 2 + 0) * CC, mrel + (li * 2 + 0) * HDD,
                                         bv + (li * 2 + 0) * kC, WtV, bVf);
    gemm128_wmma<<<kNC / 16, 256, 0, stream>>>(xcb, WtQ0, bq + (li * 2 + 0) * kC, q, nullptr, nullptr, 0);
    gemm128_wmma<<<kNV / 16, 256, 0, stream>>>(xvb, WtQ1, bq + (li * 2 + 1) * kC, qv, nullptr, nullptr, 0);
    gemm128_wmma<<<kNC / 16, 256, 0, stream>>>(xcb, WtK, bKf, krel, nullptr, nullptr, 0);
    gemm128_wmma<<<kNC / 16, 256, 0, stream>>>(xcb, WtV, bVf, vrel, nullptr, nullptr, 0);
    // cell -> cell attention
    fill_u32<<<nb(kNC * kH), 256, 0, stream>>>(menc, 0u, kNC * kH);
    fill_f32<<<nb(kNC * kH), 256, 0, stream>>>(ssum, 0.f, kNC * kH);
    fill_f32<<<nb((long)kNC * kC), 256, 0, stream>>>(agg, 0.f, kNC * kC);
    edge_logits_max<<<nb((long)kE1 * 32), 256, 0, stream>>>(q, krel, src_cc, dst_cc,
                                                            prel + (li * 2 + 0) * kH, elog, menc, kE1);
    edge_exp_sum<<<nb((long)kE1 * 4), 256, 0, stream>>>(dst_cc, menc, elog, ssum, kE1 * 4);
    edge_scatter<<<nb((long)kE1 * 32), 256, 0, stream>>>(vrel, elog, ssum, src_cc, dst_cc, agg, kE1);
    // cell -> vcell attention (relation 1 folded weights, reuse krel/vrel buffers)
    fold_weight<<<128, 128, 0, stream>>>(Wk + (li * 2 + 0) * CC, arel + (li * 2 + 1) * HDD,
                                         bk + (li * 2 + 0) * kC, WtK, bKf);
    fold_weight<<<128, 128, 0, stream>>>(Wv + (li * 2 + 0) * CC, mrel + (li * 2 + 1) * HDD,
                                         bv + (li * 2 + 0) * kC, WtV, bVf);
    gemm128_wmma<<<kNC / 16, 256, 0, stream>>>(xcb, WtK, bKf, krel, nullptr, nullptr, 0);
    gemm128_wmma<<<kNC / 16, 256, 0, stream>>>(xcb, WtV, bVf, vrel, nullptr, nullptr, 0);
    fill_u32<<<nb(kNV * kH), 256, 0, stream>>>(mvenc, 0u, kNV * kH);
    fill_f32<<<nb(kNV * kH), 256, 0, stream>>>(svsum, 0.f, kNV * kH);
    fill_f32<<<nb((long)kNV * kC), 256, 0, stream>>>(aggv, 0.f, kNV * kC);
    edge_logits_max<<<nb((long)kE2 * 32), 256, 0, stream>>>(qv, krel, src_cv, dst_cv,
                                                            prel + (li * 2 + 1) * kH, elog, mvenc, kE2);
    edge_exp_sum<<<nb((long)kE2 * 4), 256, 0, stream>>>(dst_cv, mvenc, elog, svsum, kE2 * 4);
    edge_scatter<<<nb((long)kE2 * 32), 256, 0, stream>>>(vrel, elog, svsum, src_cv, dst_cv, aggv, kE2);
    // output projection + fused skip update
    gelu_cvt_bf16<<<nb((long)kNC * kC), 256, 0, stream>>>(agg, xcb, kNC * kC);
    gelu_cvt_bf16<<<nb((long)kNV * kC), 256, 0, stream>>>(aggv, xvb, kNV * kC);
    transpose_cvt<<<128, 128, 0, stream>>>(Wa + (li * 2 + 0) * CC, WtA0);
    transpose_cvt<<<128, 128, 0, stream>>>(Wa + (li * 2 + 1) * CC, WtA1);
    gemm128_wmma<<<kNC / 16, 256, 0, stream>>>(xcb, WtA0, ba + (li * 2 + 0) * kC, xc, xc, skip + li * 2 + 0, 1);
    gemm128_wmma<<<kNV / 16, 256, 0, stream>>>(xvb, WtA1, ba + (li * 2 + 1) * kC, xvv, xvv, skip + li * 2 + 1, 1);
    // out-conv (scalar head) on virtual edges
    node_dot2<<<nb((long)kNC * 32), 256, 0, stream>>>(xc, Wk_o + li * kC, bk_o + li,
                                                      Wv_o + li * kC, bv_o + li, ko, vo, kNC);
    node_dot2<<<nb((long)kNV * 32), 256, 0, stream>>>(xvv, Wq_o + li * kC, bq_o + li,
                                                      nullptr, nullptr, qo, nullptr, kNV);
    fill_u32<<<1, 256, 0, stream>>>(moenc, 0u, kNV);
    fill_f32<<<1, 256, 0, stream>>>(sosum, 0.f, kNV);
    fill_f32<<<1, 256, 0, stream>>>(aggo, 0.f, kNV);
    oedge_logits_max<<<nb(kE2), 256, 0, stream>>>(qo, ko, src_cv, dst_cv, arel_o, prel_o, li, elogo, moenc, kE2);
    oedge_exp_sum<<<nb(kE2), 256, 0, stream>>>(dst_cv, moenc, elogo, sosum, kE2);
    oedge_scatter<<<nb(kE2), 256, 0, stream>>>(vo, elogo, sosum, mrel_o, li, src_cv, dst_cv, aggo, kE2);
    outs_k<<<1, 256, 0, stream>>>(aggo, Wa_o, ba_o, li, outsBuf);
  }
  final_k<<<1, 64, 0, stream>>>(outsBuf, fc_W, fc_b, mlp_W1, mlp_b1, mlp_W2, mlp_b2, (float*)d_out);
}
